// WindowAttention_78168404787406
// MI455X (gfx1250) — matrix operand order repaired and checked by reading, not compiled, not run on hardware
//
#include <hip/hip_runtime.h>
#include <cmath>

typedef __attribute__((ext_vector_type(16))) _Float16 v16h;
typedef __attribute__((ext_vector_type(8)))  _Float16 v8h;
typedef __attribute__((ext_vector_type(8)))  float    v8f;

#define TOK   49      // tokens per window
#define NPAD  64      // padded token dim (4 x 16)
#define CH    96      // channels
#define HEADS 3
#define HD    32      // head dim
#define NWIN  64      // windows per image

// ---- dynamic-LDS layout (byte offsets), phase-overlaid ----
// Phase 1:  XS   f16[64][96]                 @ 0       (12288)
//           WQs  f16[3][18][32][16] (B-frag) @ 12288   (55296)
//           Q    f16[3][64][32] (row major)  @ 67584   (12288)
//           Ks   f16[3][4][32][16] (B-frag)  @ 79872   (12288)
//           Vs   f16[3][2][2][32][16](B-frag)@ 92160   (12288)
// Phase 2:  S    f32[3][64][64]              @ 0       (49152)  (over XS+WQs)
//           P    f16[3][64][64]              @ 67584   (24576)  (over Q+Ks; Vs kept)
// Phase 3+: O1   f16[64][96]                 @ 0       (12288)
//           PWs  f16[3][6][32][16] (B-frag)  @ 12288   (18432)
#define SM_X    0
#define SM_WQ   12288
#define SM_Q    67584
#define SM_K    79872
#define SM_V    92160
#define SM_S    0
#define SM_P    67584
#define SM_O1   0
#define SM_PW   12288
#define SMEM_BYTES 104448

// ---------- fragment loaders (contiguous k order: lane half h holds k = 16h..16h+15) ----------

// A-matrix 16x32 f16 from row-major LDS: per lane, K = {16h..16h+15}
// -> two contiguous 16B loads (2 x ds_load_b128).
__device__ __forceinline__ v16h load_A(const _Float16* base, int ld, int row0, int k0, int lane) {
    const int hlf = lane >> 4;
    const _Float16* rp = base + (row0 + (lane & 15)) * ld + k0 + hlf * 16;
    v8h lo = *(const v8h*)rp;
    v8h hi = *(const v8h*)(rp + 8);
    return __builtin_shufflevector(lo, hi, 0,1,2,3,4,5,6,7,8,9,10,11,12,13,14,15);
}

// B-matrix 32x16 f16 stored pre-swizzled in fragment order [lane][16 halves]:
// element e of lane l == B[16*(l>>4)+e][l&15]  -> one 32B contiguous load.
__device__ __forceinline__ v16h load_frag32(const _Float16* frag, int lane) {
    return *(const v16h*)(frag + lane * 16);
}

__device__ __forceinline__ v8f wmma32(v16h a, v16h b, v8f c) {
    return __builtin_amdgcn_wmma_f32_16x16x32_f16(false, a, false, b, (short)0, c, false, false);
}

// ---------------- pre-kernel: BM[w][h][64][64] = rpb[rel_index] + mask (zero-padded) ----------------
__global__ __launch_bounds__(256)
void build_bm(const float* __restrict__ mask, const float* __restrict__ rpb,
              const int* __restrict__ rel_index, float* __restrict__ bm)
{
    const int idx = blockIdx.x * 256 + threadIdx.x;          // NWIN*HEADS*64*64
    if (idx >= NWIN * HEADS * NPAD * NPAD) return;
    const int m = idx & 63;
    const int n = (idx >> 6) & 63;
    const int h = (idx >> 12) % HEADS;
    const int w = (idx >> 12) / HEADS;
    float v = 0.0f;
    if (n < TOK && m < TOK)
        v = rpb[rel_index[n * TOK + m] * HEADS + h] + mask[w * TOK * TOK + n * TOK + m];
    bm[idx] = v;
}

__global__ __launch_bounds__(256)
void winattn_fused(const float* __restrict__ x,
                   const float* __restrict__ mask,
                   const float* __restrict__ qkv_w,
                   const float* __restrict__ qkv_b,
                   const float* __restrict__ proj_w,
                   const float* __restrict__ proj_b,
                   const float* __restrict__ rpb,
                   const int*   __restrict__ rel_index,
                   const float* __restrict__ bm,          // may be null -> gather fallback
                   float* __restrict__ out)
{
    extern __shared__ __align__(32) char smem[];
    const int tid  = threadIdx.x;
    const int lane = tid & 31;
    // wave index is wave-uniform: pin it in an SGPR so tile math / s-branches go scalar
    const int wave = __builtin_amdgcn_readfirstlane(tid >> 5);
    const int bnw  = blockIdx.x;
    const int win  = bnw & (NWIN - 1);

    _Float16* XS  = (_Float16*)(smem + SM_X);
    _Float16* WQs = (_Float16*)(smem + SM_WQ);
    _Float16* Q   = (_Float16*)(smem + SM_Q);
    _Float16* Ks  = (_Float16*)(smem + SM_K);
    _Float16* Vs  = (_Float16*)(smem + SM_V);

    // ---------------- Phase 0: stage x and W_qkv^T (fragment-swizzled), 8-half vector chunks ----------------
    {
        const float* xw = x + (size_t)bnw * TOK * CH;
        for (int c8 = tid; c8 < NPAD * CH / 8; c8 += 256) {
            const int i = c8 * 8, r = i / CH, c = i % CH;
            v8h vv;
            if (r < TOK) {
                const float* src = xw + r * CH + c;
#pragma unroll
                for (int e = 0; e < 8; ++e) vv[e] = (_Float16)src[e];
            } else {
#pragma unroll
                for (int e = 0; e < 8; ++e) vv[e] = (_Float16)0.0f;
            }
            *(v8h*)(XS + i) = vv;
        }
        // WQs fragment (kt,nt): lane l, element e -> W^T[kt*32+16*(l>>4)+e][nt*16+(l&15)]
        for (int c8 = tid; c8 < 27648 / 8; c8 += 256) {
            const int i = c8 * 8;
            const int e0 = i & 15, ln = (i >> 4) & 31, f = i >> 9;   // f = kt*18+nt
            const int nt = f % 18, kt = f / 18;
            const int cc = kt * 32 + (ln >> 4) * 16 + e0;
            const int o  = nt * 16 + (ln & 15);
            const float* src = qkv_w + o * CH + cc;
            v8h vv;
#pragma unroll
            for (int e = 0; e < 8; ++e) vv[e] = (_Float16)src[e];
            *(v8h*)(WQs + i) = vv;
        }
    }
    __syncthreads();

    // ---------------- Phase 1: QKV = X(64x96) @ W^T(96x288); A reused across 9 N-tiles ----------------
    const float qscale = 0.17677669529663687f;   // 32^-0.5
    {
        const int mt  = wave & 3;
        const int ntb = (wave >> 2) * 9;
        v16h a0 = load_A(XS, CH, mt * 16,  0, lane);
        v16h a1 = load_A(XS, CH, mt * 16, 32, lane);
        v16h a2 = load_A(XS, CH, mt * 16, 64, lane);
        for (int j = 0; j < 9; ++j) {
            const int nt = ntb + j;                          // scalar
            v8f acc = {};
            acc = wmma32(a0, load_frag32(WQs + (0 * 18 + nt) * 512, lane), acc);
            acc = wmma32(a1, load_frag32(WQs + (1 * 18 + nt) * 512, lane), acc);
            acc = wmma32(a2, load_frag32(WQs + (2 * 18 + nt) * 512, lane), acc);
            const int o = nt * 16 + (lane & 15);
            const int s = nt / 6, h = (nt >> 1) % 3;         // scalar -> s_cbranch below
            const int d = ((nt & 1) << 4) | (lane & 15);
            const float bia = qkv_b[o];
            if (s == 0) {                 // Q: row-major [h][64][32]
                _Float16* dst = Q + (h * 64 + mt * 16 + (lane >> 4) * 8) * 32 + d;
#pragma unroll
                for (int r = 0; r < 8; ++r) dst[r * 32] = (_Float16)((acc[r] + bia) * qscale);
            } else if (s == 1) {          // K: B-fragment (h, m-tile); n=token, K-dim=d
                _Float16* dst = Ks + (h * 4 + mt) * 512 + ((d >> 4) * 16 + (lane >> 4) * 8) * 16 + (d & 15);
#pragma unroll
                for (int r = 0; r < 8; ++r) dst[r * 16] = (_Float16)(acc[r] + bia);
            } else {                      // V: B-fragment (h, kt=m/32, dt=d/16); e = m&15 -> vector store
                v8h vv;
#pragma unroll
                for (int r = 0; r < 8; ++r) vv[r] = (_Float16)(acc[r] + bia);
                _Float16* dst = Vs + (((h * 2 + (mt >> 1)) * 2 + (d >> 4)) * 512)
                                   + (((mt & 1) * 16 + (d & 15)) * 16) + ((lane >> 4) * 8);
                *(v8h*)dst = vv;
            }
        }
    }
    __syncthreads();

    // ---------------- Phase 2: S = Q @ K^T + (rpb + mask) ----------------
    float* S = (float*)(smem + SM_S);            // [3][64][64]
    {
        const float* mw  = mask + (size_t)win * TOK * TOK;
        const float* bmw = bm ? bm + ((size_t)win * HEADS) * NPAD * NPAD : nullptr;
        for (int t = wave; t < 48; t += 8) {     // 3 heads x 4 mt x 4 nt (all scalar indices)
            const int h = t >> 4, tt = t & 15, mt = tt & 3, nt = tt >> 2;
            v16h a = load_A(Q + h * 2048, HD, mt * 16, 0, lane);
            v16h b = load_frag32(Ks + (h * 4 + nt) * 512, lane);
            v8f acc = {};
            acc = wmma32(a, b, acc);
            const int mcol = nt * 16 + (lane & 15);
#pragma unroll
            for (int r = 0; r < 8; ++r) {
                const int nrow = mt * 16 + (lane >> 4) * 8 + r;
                float v = 0.0f;
                if (nrow < TOK && mcol < TOK) {
                    if (bmw) v = acc[r] + bmw[(h * 64 + nrow) * 64 + mcol];
                    else     v = acc[r] + rpb[rel_index[nrow * TOK + mcol] * HEADS + h]
                                        + mw[nrow * TOK + mcol];
                }
                S[(h * 64 + nrow) * 64 + mcol] = v;
            }
        }
    }
    __syncthreads();

    // ---------------- Phase 2b: sign(x) * softmax(|x|) per row -> P (f16) ----------------
    _Float16* P = (_Float16*)(smem + SM_P);      // [3][64][64]
    if (tid < HEADS * TOK) {
        const int h = tid / TOK, n = tid % TOK;
        float* row = S + (h * 64 + n) * 64;
        float amax = 0.0f;
        for (int m = 0; m < TOK; ++m) amax = fmaxf(amax, fabsf(row[m]));
        float sum = 0.0f;
        for (int m = 0; m < TOK; ++m) {
            const float s0 = row[m];
            const float e  = __expf(fabsf(s0) - amax);
            sum += e;
            row[m] = (s0 > 0.0f) ? e : ((s0 < 0.0f) ? -e : 0.0f);
        }
        const float inv = 1.0f / sum;
        _Float16* pr = P + (h * 64 + n) * 64;
        for (int m = 0; m < TOK; ++m)  pr[m] = (_Float16)(row[m] * inv);
        for (int m = TOK; m < 64; ++m) pr[m] = (_Float16)0.0f;
    } else if (tid < HEADS * TOK + HEADS * (NPAD - TOK)) {
        const int z = tid - HEADS * TOK;
        const int h = z / (NPAD - TOK), n = TOK + z % (NPAD - TOK);
        _Float16* pr = P + (h * 64 + n) * 64;
        for (int m = 0; m < 64; ++m) pr[m] = (_Float16)0.0f;
    }
    __syncthreads();

    // ---------------- Phase 3: O1 = P @ V  (+ stage proj_w^T fragments) ----------------
    _Float16* O1  = (_Float16*)(smem + SM_O1);   // [64][96]
    _Float16* PWs = (_Float16*)(smem + SM_PW);   // fragments [3][6][32][16]
    for (int c8 = tid; c8 < 9216 / 8; c8 += 256) {
        const int i = c8 * 8;
        const int e0 = i & 15, ln = (i >> 4) & 31, f = i >> 9;    // f = kt*6+nt
        const int nt = f % 6, kt = f / 6;
        const int cc = kt * 32 + (ln >> 4) * 16 + e0;
        const int o  = nt * 16 + (ln & 15);
        const float* src = proj_w + o * CH + cc;
        v8h vv;
#pragma unroll
        for (int e = 0; e < 8; ++e) vv[e] = (_Float16)src[e];
        *(v8h*)(PWs + i) = vv;
    }
    for (int t = wave; t < 24; t += 8) {         // 3 heads x 4 mt x 2 dt (scalar indices)
        const int h = t >> 3, tt = t & 7, mt = tt & 3, dt = tt >> 2;
        const _Float16* Ph = P + h * 4096;
        v8f acc = {};
#pragma unroll
        for (int kt = 0; kt < 2; ++kt) {
            v16h a = load_A(Ph, 64, mt * 16, kt * 32, lane);
            v16h b = load_frag32(Vs + ((h * 2 + kt) * 2 + dt) * 512, lane);
            acc = wmma32(a, b, acc);
        }
        const int col = h * HD + dt * 16 + (lane & 15);
#pragma unroll
        for (int r = 0; r < 8; ++r) {
            const int nrow = mt * 16 + (lane >> 4) * 8 + r;
            O1[nrow * CH + col] = (_Float16)acc[r];
        }
    }
    __syncthreads();

    // ---------------- Phase 4: out = O1 @ proj_w^T + proj_b -> HBM; A reused across 3 N-tiles ----------------
    {
        float* ow = out + (size_t)bnw * TOK * CH;
        const int mt  = wave & 3;
        const int ntb = (wave >> 2) * 3;
        v16h a0 = load_A(O1, CH, mt * 16,  0, lane);
        v16h a1 = load_A(O1, CH, mt * 16, 32, lane);
        v16h a2 = load_A(O1, CH, mt * 16, 64, lane);
        for (int j = 0; j < 3; ++j) {
            const int nt = ntb + j;
            v8f acc = {};
            acc = wmma32(a0, load_frag32(PWs + (0 * 6 + nt) * 512, lane), acc);
            acc = wmma32(a1, load_frag32(PWs + (1 * 6 + nt) * 512, lane), acc);
            acc = wmma32(a2, load_frag32(PWs + (2 * 6 + nt) * 512, lane), acc);
            const int o = nt * 16 + (lane & 15);
            const float bia = proj_b[o];
#pragma unroll
            for (int r = 0; r < 8; ++r) {
                const int nrow = mt * 16 + (lane >> 4) * 8 + r;
                if (nrow < TOK) ow[nrow * CH + o] = acc[r] + bia;
            }
        }
    }
}

extern "C" void kernel_launch(void* const* d_in, const int* in_sizes, int n_in,
                              void* d_out, int out_size, void* d_ws, size_t ws_size,
                              hipStream_t stream) {
    const float* x      = (const float*)d_in[0];
    const float* mask   = (const float*)d_in[1];
    const float* qkv_w  = (const float*)d_in[2];
    const float* qkv_b  = (const float*)d_in[3];
    const float* proj_w = (const float*)d_in[4];
    const float* proj_b = (const float*)d_in[5];
    const float* rpb    = (const float*)d_in[6];
    const int*   relidx = (const int*)d_in[7];
    float* out = (float*)d_out;
    (void)in_sizes; (void)n_in; (void)out_size;

    const size_t bm_bytes = (size_t)NWIN * HEADS * NPAD * NPAD * sizeof(float);
    float* bm = (d_ws != nullptr && ws_size >= bm_bytes) ? (float*)d_ws : nullptr;
    if (bm) {
        const int total = NWIN * HEADS * NPAD * NPAD;
        build_bm<<<(total + 255) / 256, 256, 0, stream>>>(mask, rpb, relidx, bm);
    }
    winattn_fused<<<4096, 256, SMEM_BYTES, stream>>>(
        x, mask, qkv_w, qkv_b, proj_w, proj_b, rpb, relidx, bm, out);
}
